// SlidingWindowAttention_36790689858315
// MI455X (gfx1250) — matrix operand order repaired and checked by reading, not compiled, not run on hardware
//
#include <hip/hip_runtime.h>
#include <hip/hip_bf16.h>

// ---------------------------------------------------------------------------
// Sliding-window attention, MI455X (gfx1250, wave32, WMMA bf16 + TDM path).
// Pipeline: cvt(fp32->bf16) -> QKV WMMA GEMMs -> flash attention (WMMA) ->
//           output-projection WMMA GEMM (fp32 out).
// All WMMA fragments are fed by contiguous 16B ds_load_b128 (B and V are
// staged transposed in LDS via v_perm_b32 pair-packing); A/K tiles staged by
// the Tensor Data Mover. One-shot streams use non-temporal cache hints.
// ---------------------------------------------------------------------------

typedef __bf16 bf16_t;
typedef __attribute__((ext_vector_type(16))) __bf16 v16bf;
typedef __attribute__((ext_vector_type(8)))  __bf16 v8bf;
typedef __attribute__((ext_vector_type(8)))  float  v8f;
typedef __attribute__((ext_vector_type(4)))  float  v4f;

#define TT 2048
#define CC 1024
#define HH 16
#define DD 64
#define WIN 256
#define BB 2
#define MM (BB*TT)   // 4096 token rows

union FragA { v16bf v; v8bf h[2]; };
union V8U   { v8bf v; unsigned u[4]; };

__device__ __forceinline__ v8f wmma_bf16(v16bf a, v16bf b, v8f c) {
  // emits v_wmma_f32_16x16x32_bf16
  return __builtin_amdgcn_wmma_f32_16x16x32_bf16(
      false, a, false, b, (short)0, c, false, false);
}

// pack element e of two bf16x8 rows into one dword {r1.e, r0.e} via v_perm_b32
__device__ __forceinline__ unsigned pack_pair(const V8U& r0, const V8U& r1, int e) {
  unsigned sel = (e & 1) ? 0x07060302u : 0x05040100u;
  return __builtin_amdgcn_perm(r1.u[e >> 1], r0.u[e >> 1], sel);
}

// ---------------------------------------------------------------------------
// Tensor Data Mover (gfx1250): D# per CDNA5 ISA ch.8, 6-arg builtin:
//   (u32x4 g0, i32x8 g1, i32x4 g2, i32x4 g3, i32x8 extra, i32 cpol)
// ---------------------------------------------------------------------------
#if defined(__gfx1250__) && __has_builtin(__builtin_amdgcn_tensor_load_to_lds) \
    && __has_builtin(__builtin_amdgcn_s_wait_tensorcnt)
#define HAVE_TDM 1
#else
#define HAVE_TDM 0
#endif

#if HAVE_TDM
typedef unsigned int u32x4 __attribute__((ext_vector_type(4)));
typedef int          i32x4 __attribute__((ext_vector_type(4)));
typedef int          i32x8 __attribute__((ext_vector_type(8)));

__device__ __forceinline__ unsigned lds_offset(const void* p) {
  return (unsigned)(size_t)(__attribute__((address_space(3))) const void*)p;
}

// 2D bf16 tile load, global(row-major) -> LDS(row-major, packed).
// tile_w elems contiguous per row, tile_h rows, row stride = stride_e elems.
__device__ __forceinline__ void tdm_load_2d_bf16(
    unsigned lds_off, const void* gptr,
    unsigned tile_w, unsigned tile_h, unsigned stride_e) {
  unsigned long long ga = (unsigned long long)(size_t)gptr;
  u32x4 g0;
  g0[0] = 1u;                                              // count=1, user D#
  g0[1] = lds_off;                                         // lds_addr (bytes)
  g0[2] = (unsigned)(ga & 0xFFFFFFFFull);                  // global_addr lo
  g0[3] = (unsigned)((ga >> 32) & 0x1FFFFFFull) | (2u << 30); // hi | type=2
  i32x8 g1;
  g1[0] = (int)(1u << 16);                                 // data_size=2B
  g1[1] = (int)((tile_w & 0xFFFFu) << 16);                 // tensor_dim0 lo16
  g1[2] = (int)((tile_w >> 16) | ((tile_h & 0xFFFFu) << 16)); // dim0 hi/dim1 lo
  g1[3] = (int)((tile_h >> 16) | (tile_w << 16));          // dim1 hi | tile_dim0
  g1[4] = (int)(tile_h & 0xFFFFu);                         // tile_dim1, tile_dim2=0
  g1[5] = (int)stride_e;                                   // tensor_dim0_stride
  g1[6] = 0;
  g1[7] = 0;
  i32x4 z4 = {0, 0, 0, 0};
  i32x8 z8 = {0, 0, 0, 0, 0, 0, 0, 0};
  __builtin_amdgcn_tensor_load_to_lds(g0, g1, z4, z4, z8, 0);
}
#endif

// ---------------------------------------------------------------------------
// fp32 -> bf16 conversion (non-temporal source reads: zero reuse)
// ---------------------------------------------------------------------------
__global__ __launch_bounds__(256) void cvt_f32_bf16(
    const float* __restrict__ src, bf16_t* __restrict__ dst, int n) {
  int i = (blockIdx.x * blockDim.x + threadIdx.x) * 4;
  if (i + 3 < n) {
    v4f f = __builtin_nontemporal_load((const v4f*)(src + i));
    dst[i + 0] = (bf16_t)f.x;
    dst[i + 1] = (bf16_t)f.y;
    dst[i + 2] = (bf16_t)f.z;
    dst[i + 3] = (bf16_t)f.w;
  } else {
    for (; i < n; ++i) dst[i] = (bf16_t)src[i];
  }
}

// ---------------------------------------------------------------------------
// Tiled WMMA GEMM:  out = A(MxK,bf16) @ B(KxN,bf16) + bias
// 256 threads = 8 waves; block tile 64(M) x 64(N); K-step 64.
// Wave (mi,ni) -> rows mi*16, cols ni*32 (two 16x16 C tiles), 4 WMMA/stage.
// A staged row-major by TDM; B staged TRANSPOSED (v_perm pair-pack, b32
// stores) so B-fragments are two contiguous ds_load_b128 per tile.
// mode 0: fp32 row-major out [M,N] (non-temporal);  mode 1: bf16 -> [B,H,T,D]
// ---------------------------------------------------------------------------
#define BT_STRIDE 72   // transposed-B LDS row stride in elems (144B, 16B mult)

__global__ __launch_bounds__(256) void gemm_wmma(
    const bf16_t* __restrict__ A, const bf16_t* __restrict__ Bm,
    const float* __restrict__ bias, void* __restrict__ out,
    int M, int N, int K, int mode) {
  __shared__ bf16_t ldsA[64 * 64];
  __shared__ bf16_t ldsBt[64 * BT_STRIDE];

  const int tid  = threadIdx.x;
  const int wave = tid >> 5, lane = tid & 31;
  const int mi   = wave >> 1, ni = wave & 1;
  const int Mb   = blockIdx.x * 64, Nb = blockIdx.y * 64;
  const int m    = lane & 15, half = lane >> 4;

  // staging coordinates
  const int arow = tid >> 2, ac16 = (tid & 3) * 16;   // A: 64 rows x 4 chunks
  const int kp   = (tid >> 3) * 2;                    // B: even k of a k-pair
  const int bc8  = (tid & 7) * 8;                     // B: 8-col chunk

  v8f c0 = {}, c1 = {};
  for (int kk = 0; kk < K; kk += 64) {
    __syncthreads();
#if HAVE_TDM
    if (wave == 0) {
      tdm_load_2d_bf16(lds_offset(ldsA), A + (size_t)Mb * K + kk, 64, 64,
                       (unsigned)K);
    }
#else
    {
      const bf16_t* gA = A + (size_t)(Mb + arow) * K + kk + ac16;
      *(v8bf*)(ldsA + arow * 64 + ac16)     = *(const v8bf*)(gA);
      *(v8bf*)(ldsA + arow * 64 + ac16 + 8) = *(const v8bf*)(gA + 8);
    }
#endif
    {
      // stage B 64x64 tile transposed: v_perm-pack (k,k+1) pairs, b32 stores
      const bf16_t* gB = Bm + (size_t)(kk + kp) * N + Nb + bc8;
      V8U r0, r1;
      r0.v = *(const v8bf*)(gB);
      r1.v = *(const v8bf*)(gB + N);
      if (kk + 64 < K) __builtin_prefetch(gB + (size_t)64 * N, 0, 1);
#pragma unroll
      for (int e = 0; e < 8; ++e) {
        *(unsigned*)(ldsBt + (bc8 + e) * BT_STRIDE + kp) = pack_pair(r0, r1, e);
      }
    }
#if HAVE_TDM
    if (wave == 0) __builtin_amdgcn_s_wait_tensorcnt(0);
#endif
    __syncthreads();

#pragma unroll
    for (int ks = 0; ks < 2; ++ks) {
      FragA a;  // A frag: M=lane&15, K groups by lane-half
      const bf16_t* ar = ldsA + (mi * 16 + m) * 64 + ks * 32;
      a.h[0] = *(const v8bf*)(ar + half * 16);
      a.h[1] = *(const v8bf*)(ar + half * 16 + 8);
      FragA b0, b1;  // B frags from transposed LDS: 2x contiguous 16B each
      const bf16_t* br0 = ldsBt + (ni * 32 + m) * BT_STRIDE + ks * 32 + half * 16;
      const bf16_t* br1 = br0 + 16 * BT_STRIDE;
      b0.h[0] = *(const v8bf*)(br0);
      b0.h[1] = *(const v8bf*)(br0 + 8);
      b1.h[0] = *(const v8bf*)(br1);
      b1.h[1] = *(const v8bf*)(br1 + 8);
      c0 = wmma_bf16(a.v, b0.v, c0);
      c1 = wmma_bf16(a.v, b1.v, c1);
    }
  }

  // C layout: lanes 0-15 row r, lanes 16-31 row r+8; col = lane&15
  const int col0 = Nb + ni * 32 + m;
  const float bc0 = bias[col0], bc1 = bias[col0 + 16];
#pragma unroll
  for (int r = 0; r < 8; ++r) {
    const int row = Mb + mi * 16 + r + half * 8;
    const float v0 = c0[r] + bc0, v1 = c1[r] + bc1;
    if (mode == 0) {
      float* o = (float*)out + (size_t)row * N;
      __builtin_nontemporal_store(v0, o + col0);        // final output: NT
      __builtin_nontemporal_store(v1, o + col0 + 16);
    } else {
      const int bb = row >> 11, t = row & (TT - 1);
      bf16_t* ob = (bf16_t*)out;
      int h0 = col0 >> 6, d0 = col0 & (DD - 1);
      int c1i = col0 + 16;
      int h1 = c1i >> 6, d1 = c1i & (DD - 1);
      ob[(((size_t)(bb * HH + h0)) * TT + t) * DD + d0] = (bf16_t)v0;
      ob[(((size_t)(bb * HH + h1)) * TT + t) * DD + d1] = (bf16_t)v1;
    }
  }
}

// ---------------------------------------------------------------------------
// Flash attention, sliding window (j<=i, i-j<=WIN).
// Grid: (B*H, T/128). 8 waves/block; wave w owns 16 queries.
// K tile staged row-major by TDM; V tile staged TRANSPOSED (v_perm pack) so
// PV B-fragments are two contiguous ds_load_b128.
// ---------------------------------------------------------------------------
#define VT_STRIDE 40   // transposed-V LDS row stride in elems (80B, 16B mult)

__global__ __launch_bounds__(256) void attn_wmma(
    const bf16_t* __restrict__ q, const bf16_t* __restrict__ k,
    const bf16_t* __restrict__ v, bf16_t* __restrict__ y) {
  __shared__ bf16_t ldsK[32 * 64];
  __shared__ bf16_t ldsVt[64 * VT_STRIDE];
  __shared__ bf16_t ldsP[8 * 16 * 32];

  const int bh  = blockIdx.x;          // b*H + h
  const int i0w = blockIdx.y * 128;    // block's query base
  const int tid = threadIdx.x, wave = tid >> 5, lane = tid & 31;
  const int iw  = i0w + wave * 16;     // wave's query base
  const int m   = lane & 15, half = lane >> 4;
  const size_t base = (size_t)bh * TT * DD;

  // Q fragments (K-dim 64 -> two 16x32 A-fragments), loaded once
  FragA aq0, aq1;
  {
    const bf16_t* qrow = q + base + (size_t)(iw + m) * DD;
    aq0.h[0] = *(const v8bf*)(qrow + half * 16);
    aq0.h[1] = *(const v8bf*)(qrow + half * 16 + 8);
    aq1.h[0] = *(const v8bf*)(qrow + 32 + half * 16);
    aq1.h[1] = *(const v8bf*)(qrow + 32 + half * 16 + 8);
  }

  v8f acc0 = {}, acc1 = {}, acc2 = {}, acc3 = {};
  float rm[8], rs[8];
#pragma unroll
  for (int r = 0; r < 8; ++r) { rm[r] = -1e30f; rs[r] = 0.f; }

  int jstart = i0w - WIN; if (jstart < 0) jstart = 0;
  const int nsteps = (i0w + 128 - jstart) >> 5;
  bf16_t* Pl = ldsP + wave * (16 * 32);

  for (int s = 0; s < nsteps; ++s) {
    const int j0 = jstart + s * 32;
    __syncthreads();
#if HAVE_TDM
    if (wave == 0) {
      tdm_load_2d_bf16(lds_offset(ldsK), k + base + (size_t)j0 * DD, 64, 32, DD);
    }
#else
    {
      int krow = tid >> 3, kc8 = (tid & 7) * 8;
      *(v8bf*)(ldsK + krow * 64 + kc8) =
          *(const v8bf*)(k + base + (size_t)(j0 + krow) * DD + kc8);
    }
#endif
    if (tid < 128) {
      // stage V 32x64 transposed: v_perm-pack (j,j+1) pairs, b32 stores
      const int jp = (tid >> 3) * 2, dc8 = (tid & 7) * 8;
      const bf16_t* gV = v + base + (size_t)(j0 + jp) * DD + dc8;
      V8U r0, r1;
      r0.v = *(const v8bf*)(gV);
      r1.v = *(const v8bf*)(gV + DD);
#pragma unroll
      for (int e = 0; e < 8; ++e) {
        *(unsigned*)(ldsVt + (dc8 + e) * VT_STRIDE + jp) = pack_pair(r0, r1, e);
      }
    }
#if HAVE_TDM
    if (wave == 0) __builtin_amdgcn_s_wait_tensorcnt(0);
#endif
    __syncthreads();
    if (j0 > iw + 15 || j0 + 31 < iw - WIN) continue;  // outside wave's window

    // ---- scores: S = Q @ K^T, two 16-col key tiles ----
    v8f s0 = {}, s1 = {};
#pragma unroll
    for (int nt = 0; nt < 2; ++nt) {
      const bf16_t* krow = ldsK + (nt * 16 + m) * 64;  // key row, contiguous d
      FragA b0, b1;
      b0.h[0] = *(const v8bf*)(krow + half * 16);
      b0.h[1] = *(const v8bf*)(krow + half * 16 + 8);
      b1.h[0] = *(const v8bf*)(krow + 32 + half * 16);
      b1.h[1] = *(const v8bf*)(krow + 32 + half * 16 + 8);
      v8f cs = {};
      cs = wmma_bf16(aq0.v, b0.v, cs);
      cs = wmma_bf16(aq1.v, b1.v, cs);
      if (nt == 0) s0 = cs; else s1 = cs;
    }

    // ---- online softmax (fp32), mask, write P (bf16) to LDS ----
    const float scale = 0.125f;  // 1/sqrt(64)
#pragma unroll
    for (int r = 0; r < 8; ++r) {
      const int i  = iw + r + half * 8;
      const int n0 = j0 + m, n1 = j0 + 16 + m;
      float x0 = s0[r] * scale, x1 = s1[r] * scale;
      if (n0 > i || i - n0 > WIN) x0 = -1e30f;
      if (n1 > i || i - n1 > WIN) x1 = -1e30f;
      float mx = fmaxf(x0, x1);
      for (int off = 1; off < 16; off <<= 1)
        mx = fmaxf(mx, __shfl_xor(mx, off, 16));
      const float nm    = fmaxf(rm[r], mx);
      const float alpha = __expf(rm[r] - nm);
      rm[r] = nm;
      const float p0 = __expf(x0 - nm), p1 = __expf(x1 - nm);
      float ps = p0 + p1;
      for (int off = 1; off < 16; off <<= 1) ps += __shfl_xor(ps, off, 16);
      rs[r] = rs[r] * alpha + ps;
      acc0[r] *= alpha; acc1[r] *= alpha; acc2[r] *= alpha; acc3[r] *= alpha;
      Pl[(r + half * 8) * 32 + m]      = (bf16_t)p0;
      Pl[(r + half * 8) * 32 + 16 + m] = (bf16_t)p1;
    }

    // ---- PV: acc += P(16x32) @ V(32x64) ----
    FragA ap;  // A-fragment of P from per-wave LDS scratch
    ap.h[0] = *(const v8bf*)(Pl + m * 32 + half * 16);
    ap.h[1] = *(const v8bf*)(Pl + m * 32 + half * 16 + 8);
#pragma unroll
    for (int dt = 0; dt < 4; ++dt) {
      const bf16_t* vr = ldsVt + (dt * 16 + m) * VT_STRIDE + half * 16;
      FragA bv;  // V B-frag: col d, 16 consecutive keys from transposed LDS
      bv.h[0] = *(const v8bf*)(vr);
      bv.h[1] = *(const v8bf*)(vr + 8);
      v8f& a = (dt == 0) ? acc0 : (dt == 1) ? acc1 : (dt == 2) ? acc2 : acc3;
      a = wmma_bf16(ap.v, bv.v, a);
    }
  }

  // ---- normalize + store y[b*T+t, h*64+d] (bf16, token-major for out-proj)
  const int b = bh >> 4, h = bh & 15;
#pragma unroll
  for (int r = 0; r < 8; ++r) {
    const int t     = iw + r + half * 8;
    const float inv = 1.0f / rs[r];
    bf16_t* yr = y + ((size_t)b * TT + t) * CC + h * DD;
    yr[ 0 + m] = (bf16_t)(acc0[r] * inv);
    yr[16 + m] = (bf16_t)(acc1[r] * inv);
    yr[32 + m] = (bf16_t)(acc2[r] * inv);
    yr[48 + m] = (bf16_t)(acc3[r] * inv);
  }
}

// ---------------------------------------------------------------------------
// Host launch
// ---------------------------------------------------------------------------
extern "C" void kernel_launch(void* const* d_in, const int* in_sizes, int n_in,
                              void* d_out, int out_size, void* d_ws, size_t ws_size,
                              hipStream_t stream) {
  const float* x  = (const float*)d_in[0];
  const float* Wq = (const float*)d_in[1];
  const float* bq = (const float*)d_in[2];
  const float* Wk = (const float*)d_in[3];
  const float* bk = (const float*)d_in[4];
  const float* Wv = (const float*)d_in[5];
  const float* bv = (const float*)d_in[6];
  const float* Wo = (const float*)d_in[7];
  const float* bo = (const float*)d_in[8];

  // Workspace layout (bf16 buffers), total ~48 MB
  char* ws = (char*)d_ws;
  const size_t SZ_X = (size_t)MM * CC * sizeof(bf16_t);   // 8 MB
  const size_t SZ_W = (size_t)CC * CC * sizeof(bf16_t);   // 2 MB
  bf16_t* xb  = (bf16_t*)(ws);
  bf16_t* wqb = (bf16_t*)(ws + SZ_X);
  bf16_t* wkb = (bf16_t*)(ws + SZ_X + 1 * SZ_W);
  bf16_t* wvb = (bf16_t*)(ws + SZ_X + 2 * SZ_W);
  bf16_t* wob = (bf16_t*)(ws + SZ_X + 3 * SZ_W);
  bf16_t* qb  = (bf16_t*)(ws + SZ_X + 4 * SZ_W);
  bf16_t* kb  = (bf16_t*)(ws + 2 * SZ_X + 4 * SZ_W);
  bf16_t* vb  = (bf16_t*)(ws + 3 * SZ_X + 4 * SZ_W);
  bf16_t* yb  = (bf16_t*)(ws + 4 * SZ_X + 4 * SZ_W);

  const int nx = MM * CC, nw = CC * CC;
  cvt_f32_bf16<<<nx / 1024, 256, 0, stream>>>(x,  xb,  nx);
  cvt_f32_bf16<<<nw / 1024, 256, 0, stream>>>(Wq, wqb, nw);
  cvt_f32_bf16<<<nw / 1024, 256, 0, stream>>>(Wk, wkb, nw);
  cvt_f32_bf16<<<nw / 1024, 256, 0, stream>>>(Wv, wvb, nw);
  cvt_f32_bf16<<<nw / 1024, 256, 0, stream>>>(Wo, wob, nw);

  dim3 ggrid(MM / 64, CC / 64);  // 64 x 16 blocks, 64x64 tile each
  gemm_wmma<<<ggrid, 256, 0, stream>>>(xb, wqb, bq, qb, MM, CC, CC, 1);
  gemm_wmma<<<ggrid, 256, 0, stream>>>(xb, wkb, bk, kb, MM, CC, CC, 1);
  gemm_wmma<<<ggrid, 256, 0, stream>>>(xb, wvb, bv, vb, MM, CC, CC, 1);

  attn_wmma<<<dim3(BB * HH, TT / 128), 256, 0, stream>>>(qb, kb, vb, yb);

  gemm_wmma<<<ggrid, 256, 0, stream>>>(yb, wob, bo, d_out, MM, CC, CC, 0);
}
